// SelfAttention_89335319756888
// MI455X (gfx1250) — matrix operand order repaired and checked by reading, not compiled, not run on hardware
//
#include <hip/hip_runtime.h>

// ---------------------------------------------------------------------------
// Problem constants (reference: B=4, N=4096, D=1024, 3D=3072)
// ---------------------------------------------------------------------------
constexpr int Bn   = 4;
constexpr int Nn   = 4096;
constexpr int Dn   = 1024;
constexpr int TDn  = 3 * Dn;          // 3072
constexpr int Mrows = Bn * Nn;        // 16384

// Workspace layout (bytes). Total ~168 MB of bf16 staging.
constexpr size_t XBF_OFF   = 0;                                  // x bf16   [16384,1024]
constexpr size_t WQKV_OFF  = XBF_OFF  + (size_t)Mrows * Dn * 2;  // Wqkv bf16 [3072,1024]
constexpr size_t WO_OFF    = WQKV_OFF + (size_t)TDn * Dn * 2;    // Wo bf16   [1024,1024]
constexpr size_t Q_OFF     = WO_OFF   + (size_t)Dn * Dn * 2;     // q bf16    [16384,1024]
constexpr size_t K_OFF     = Q_OFF    + (size_t)Mrows * Dn * 2;
constexpr size_t V_OFF     = K_OFF    + (size_t)Mrows * Dn * 2;
constexpr size_t A_OFF     = V_OFF    + (size_t)Mrows * Dn * 2;  // attn out bf16

// ---------------------------------------------------------------------------
// WMMA types / helpers (CDNA5 wave32, v_wmma_f32_16x16x32_bf16)
// ---------------------------------------------------------------------------
typedef __attribute__((ext_vector_type(16))) __bf16        v16bf;
typedef __attribute__((ext_vector_type(8)))  float         v8f;
typedef __attribute__((ext_vector_type(4)))  unsigned int  v4u;

union FragB16 {
    v16bf          v;
    v4u            q[2];
    unsigned short u[16];
};

static __device__ __forceinline__ v8f wmma_bf16(v16bf a, v16bf b, v8f c) {
    // 8 args: (neg_a, A, neg_b, B, c_mod, C, reuse_a, reuse_b)
    return __builtin_amdgcn_wmma_f32_16x16x32_bf16(false, a, false, b,
                                                   (short)0, c, false, false);
}

static __device__ __forceinline__ unsigned short f2bf(float f) {
    unsigned int u = __float_as_uint(f);
    u += 0x7FFFu + ((u >> 16) & 1u);       // round-to-nearest-even
    return (unsigned short)(u >> 16);
}

static __device__ __forceinline__ float redmax16(float v) {
    v = fmaxf(v, __shfl_xor(v, 1));
    v = fmaxf(v, __shfl_xor(v, 2));
    v = fmaxf(v, __shfl_xor(v, 4));
    v = fmaxf(v, __shfl_xor(v, 8));
    return v;
}
static __device__ __forceinline__ float redsum16(float v) {
    v += __shfl_xor(v, 1);
    v += __shfl_xor(v, 2);
    v += __shfl_xor(v, 4);
    v += __shfl_xor(v, 8);
    return v;
}

// ---------------------------------------------------------------------------
// Kernel 1: f32 -> bf16 cast (grid-stride)
// ---------------------------------------------------------------------------
__global__ void cvt_bf16_kernel(const float* __restrict__ in,
                                unsigned short* __restrict__ out, int n) {
    int i = blockIdx.x * blockDim.x + threadIdx.x;
    int stride = gridDim.x * blockDim.x;
    for (; i < n; i += stride) out[i] = f2bf(in[i]);
}

// ---------------------------------------------------------------------------
// Kernel 2: QKV GEMM.  C[16384,3072] = Xbf @ Wqkvbf^T + bias, stored bf16 as
// separate q/k/v buffers. One wave per 16x16 output tile, K looped in 32s.
// ---------------------------------------------------------------------------
__global__ __launch_bounds__(256) void qkv_gemm_kernel(
    const unsigned short* __restrict__ xbf,
    const unsigned short* __restrict__ wbf,
    const float* __restrict__ bias,
    unsigned short* __restrict__ qo,
    unsigned short* __restrict__ ko,
    unsigned short* __restrict__ vo) {
    const int TN   = TDn / 16;                                   // 192
    const int wave = blockIdx.x * (blockDim.x >> 5) + (threadIdx.x >> 5);
    if (wave >= (Mrows / 16) * TN) return;
    const int lane = threadIdx.x & 31;
    const int col  = lane & 15;
    const int hh   = lane >> 4;
    const int tm = wave / TN, tn = wave % TN;
    const int m0 = tm * 16, n0 = tn * 16;

    const unsigned short* arow = xbf + (size_t)(m0 + col) * Dn;  // A: lane = row
    const unsigned short* brow = wbf + (size_t)(n0 + col) * Dn;  // B: lane = out-col

    v8f acc;
#pragma unroll
    for (int i = 0; i < 8; ++i) acc[i] = 0.0f;

    for (int kk = 0; kk < Dn; kk += 32) {
        FragB16 a, b;
        a.q[0] = *(const v4u*)(arow + kk + 8 * hh);
        a.q[1] = *(const v4u*)(arow + kk + 16 + 8 * hh);
        b.q[0] = *(const v4u*)(brow + kk + 8 * hh);
        b.q[1] = *(const v4u*)(brow + kk + 16 + 8 * hh);
        acc = wmma_bf16(a.v, b.v, acc);
    }

    const int j   = n0 + col;
    const float bj = bias[j];
    unsigned short* outp;
    int jo;
    if (n0 < Dn)          { outp = qo; jo = 0; }
    else if (n0 < 2 * Dn) { outp = ko; jo = Dn; }
    else                  { outp = vo; jo = 2 * Dn; }
#pragma unroll
    for (int r = 0; r < 8; ++r) {
        int row = m0 + r + 8 * hh;                                // C: reg = row
        outp[(size_t)row * Dn + (j - jo)] = f2bf(acc[r] + bj);
    }
}

// ---------------------------------------------------------------------------
// Kernel 3: fused causal attention (flash-style, full d=1024 per the bug,
// scale 1/sqrt(64)). One wave per (batch, 16-query tile, 256-wide d-slice).
// S = Q K^T via WMMA over K=1024; online softmax in C-fragment layout;
// P staged through 1KB LDS to become an A fragment; O += P V via WMMA.
// ---------------------------------------------------------------------------
__global__ __launch_bounds__(32) void attn_kernel(
    const unsigned short* __restrict__ qb,
    const unsigned short* __restrict__ kb_,
    const unsigned short* __restrict__ vb,
    unsigned short* __restrict__ ao) {
    __shared__ unsigned short Pst[16 * 32];   // per-wave P tile (16 q x 32 keys)

    const int lane   = threadIdx.x;
    const int col    = lane & 15;
    const int hh     = lane >> 4;
    const int dslice = blockIdx.x & 3;
    const int qt     = (blockIdx.x >> 2) & 255;
    const int bidx   = blockIdx.x >> 10;
    const int q0     = qt * 16;
    const int dsl0   = dslice * 256;

    const unsigned short* qrow = qb + (size_t)(bidx * Nn + q0 + col) * Dn;

    v8f o[16];
    float mrow[8], lrow[8];
#pragma unroll
    for (int t2 = 0; t2 < 16; ++t2)
#pragma unroll
        for (int i = 0; i < 8; ++i) o[t2][i] = 0.0f;
#pragma unroll
    for (int r = 0; r < 8; ++r) { mrow[r] = -3.0e38f; lrow[r] = 0.0f; }

    const int nkb = (q0 + 47) >> 5;           // causal: key blocks of 32
    for (int kblk = 0; kblk < nkb; ++kblk) {
        const int kb32 = kblk * 32;
        const unsigned short* kr0 = kb_ + (size_t)(bidx * Nn + kb32 + col) * Dn;
        const unsigned short* kr1 = kr0 + (size_t)16 * Dn;
        // speculative prefetch of next key block (global_prefetch_b8)
        if (kblk + 1 < nkb) __builtin_prefetch(kr0 + (size_t)32 * Dn, 0, 1);

        v8f s0, s1;
#pragma unroll
        for (int i = 0; i < 8; ++i) { s0[i] = 0.0f; s1[i] = 0.0f; }

        for (int kk = 0; kk < Dn; kk += 32) {
            FragB16 a, b0, b1;
            a.q[0]  = *(const v4u*)(qrow + kk + 8 * hh);
            a.q[1]  = *(const v4u*)(qrow + kk + 16 + 8 * hh);
            b0.q[0] = *(const v4u*)(kr0 + kk + 8 * hh);
            b0.q[1] = *(const v4u*)(kr0 + kk + 16 + 8 * hh);
            b1.q[0] = *(const v4u*)(kr1 + kk + 8 * hh);
            b1.q[1] = *(const v4u*)(kr1 + kk + 16 + 8 * hh);
            s0 = wmma_bf16(a.v, b0.v, s0);
            s1 = wmma_bf16(a.v, b1.v, s1);
        }

        // scale (1/sqrt(d_head)=0.125) + causal mask on the diagonal block
        const bool last = (kblk == nkb - 1);
#pragma unroll
        for (int r = 0; r < 8; ++r) {
            float x0 = s0[r] * 0.125f;
            float x1 = s1[r] * 0.125f;
            if (last) {
                int qq = q0 + r + 8 * hh;
                if (kb32 + col > qq)      x0 = -3.0e38f;
                if (kb32 + 16 + col > qq) x1 = -3.0e38f;
            }
            s0[r] = x0;
            s1[r] = x1;
        }

        // online softmax, per C-fragment row (row = r + 8*hh, cols across lanes)
#pragma unroll
        for (int r = 0; r < 8; ++r) {
            float tmax = redmax16(fmaxf(s0[r], s1[r]));
            float mn   = fmaxf(mrow[r], tmax);
            float sc   = __expf(mrow[r] - mn);
            float p0   = __expf(s0[r] - mn);
            float p1   = __expf(s1[r] - mn);
            float su   = redsum16(p0 + p1);
            lrow[r] = lrow[r] * sc + su;
            mrow[r] = mn;
#pragma unroll
            for (int t2 = 0; t2 < 16; ++t2) o[t2][r] *= sc;
            int row = r + 8 * hh;
            Pst[row * 32 + col]      = f2bf(p0);
            Pst[row * 32 + 16 + col] = f2bf(p1);
        }
        asm volatile("s_wait_dscnt 0" ::: "memory");

        // reload P as an A fragment (lane = row, half-lane K offset)
        FragB16 pa;
        pa.q[0] = *(const v4u*)(&Pst[col * 32 + 16 * hh]);
        pa.q[1] = *(const v4u*)(&Pst[col * 32 + 16 * hh + 8]);

        // O += P * V for this wave's 256-wide d slice (16 column tiles)
        const unsigned short* vbase =
            vb + (size_t)(bidx * Nn + kb32 + 16 * hh) * Dn + dsl0 + col;
#pragma unroll
        for (int t2 = 0; t2 < 16; ++t2) {
            FragB16 bvf;
#pragma unroll
            for (int e = 0; e < 16; ++e)
                bvf.u[e] = vbase[(size_t)e * Dn + t2 * 16];
            o[t2] = wmma_bf16(pa.v, bvf.v, o[t2]);
        }
    }

    // epilogue: normalize by l and store bf16
#pragma unroll
    for (int r = 0; r < 8; ++r) {
        float inv = 1.0f / lrow[r];
        int row   = q0 + r + 8 * hh;
#pragma unroll
        for (int t2 = 0; t2 < 16; ++t2) {
            ao[(size_t)(bidx * Nn + row) * Dn + dsl0 + t2 * 16 + col] =
                f2bf(o[t2][r] * inv);
        }
    }
}

// ---------------------------------------------------------------------------
// Kernel 4: output projection with the reference's scramble folded into the
// A-fragment gather: scr(b,y,z) = attn[b, 1024*(y&3)+z, y>>2]  (exact, since
// flat=1024y+z never crosses a 4096 boundary). out = scr @ Wo^T + bias (f32).
// ---------------------------------------------------------------------------
__global__ __launch_bounds__(256) void proj_kernel(
    const unsigned short* __restrict__ attn,
    const unsigned short* __restrict__ wo,
    const float* __restrict__ bias,
    float* __restrict__ out) {
    const int TN   = Dn / 16;                                    // 64
    const int wave = blockIdx.x * (blockDim.x >> 5) + (threadIdx.x >> 5);
    if (wave >= (Mrows / 16) * TN) return;
    const int lane = threadIdx.x & 31;
    const int col  = lane & 15;
    const int hh   = lane >> 4;
    const int tm = wave / TN, tn = wave % TN;
    const int gr0 = tm * 16, j0 = tn * 16;

    const int gr = gr0 + col;                 // global row for the A gather
    const int bb = gr >> 12;                  // /4096
    const int y  = gr & (Nn - 1);
    const unsigned short* abase =
        attn + ((size_t)bb * Nn + (size_t)1024 * (y & 3)) * Dn + (y >> 2);
    const unsigned short* brow = wo + (size_t)(j0 + col) * Dn;

    v8f acc;
#pragma unroll
    for (int i = 0; i < 8; ++i) acc[i] = 0.0f;

    for (int kk = 0; kk < Dn; kk += 32) {
        FragB16 a, b;
#pragma unroll
        for (int e = 0; e < 8; ++e)
            a.u[e] = abase[(size_t)(kk + 8 * hh + e) * Dn];
#pragma unroll
        for (int e = 0; e < 8; ++e)
            a.u[8 + e] = abase[(size_t)(kk + 16 + 8 * hh + e) * Dn];
        b.q[0] = *(const v4u*)(brow + kk + 8 * hh);
        b.q[1] = *(const v4u*)(brow + kk + 16 + 8 * hh);
        acc = wmma_bf16(a.v, b.v, acc);
    }

    const float bj = bias[j0 + col];
#pragma unroll
    for (int r = 0; r < 8; ++r)
        out[(size_t)(gr0 + r + 8 * hh) * Dn + j0 + col] = acc[r] + bj;
}

// ---------------------------------------------------------------------------
// Launch
// ---------------------------------------------------------------------------
extern "C" void kernel_launch(void* const* d_in, const int* in_sizes, int n_in,
                              void* d_out, int out_size, void* d_ws, size_t ws_size,
                              hipStream_t stream) {
    (void)in_sizes; (void)n_in; (void)out_size; (void)ws_size;
    const float* x    = (const float*)d_in[0];
    const float* wqkv = (const float*)d_in[1];
    const float* bqkv = (const float*)d_in[2];
    const float* wo   = (const float*)d_in[3];
    const float* bo   = (const float*)d_in[4];

    char* ws = (char*)d_ws;
    unsigned short* xbf    = (unsigned short*)(ws + XBF_OFF);
    unsigned short* wqkvbf = (unsigned short*)(ws + WQKV_OFF);
    unsigned short* wobf   = (unsigned short*)(ws + WO_OFF);
    unsigned short* qbf    = (unsigned short*)(ws + Q_OFF);
    unsigned short* kbf    = (unsigned short*)(ws + K_OFF);
    unsigned short* vbf    = (unsigned short*)(ws + V_OFF);
    unsigned short* abf    = (unsigned short*)(ws + A_OFF);

    cvt_bf16_kernel<<<4096, 256, 0, stream>>>(x,    xbf,    Mrows * Dn);
    cvt_bf16_kernel<<<2048, 256, 0, stream>>>(wqkv, wqkvbf, TDn * Dn);
    cvt_bf16_kernel<<<1024, 256, 0, stream>>>(wo,   wobf,   Dn * Dn);

    // (16384/16)*(3072/16) = 196608 tiles, 8 waves/block
    qkv_gemm_kernel<<<24576, 256, 0, stream>>>(xbf, wqkvbf, bqkv, qbf, kbf, vbf);

    // 4 batches * 256 query tiles * 4 d-slices = 4096 single-wave blocks
    attn_kernel<<<4096, 32, 0, stream>>>(qbf, kbf, vbf, abf);

    // (16384/16)*(1024/16) = 65536 tiles, 8 waves/block
    proj_kernel<<<8192, 256, 0, stream>>>(abf, wobf, bo, (float*)d_out);
}
